// MCKRL_19421842113025
// MI455X (gfx1250) — matrix operand order repaired and checked by reading, not compiled, not run on hardware
//
#include <hip/hip_runtime.h>

#define N_ENT   14951
#define N_REL   1345
#define N_NODES 120000
#define N_EDGES 480000

typedef __attribute__((ext_vector_type(16))) __bf16 v16bf;
typedef __attribute__((ext_vector_type(8)))  __bf16 v8bf;
typedef __attribute__((ext_vector_type(2)))  __bf16 v2bf;
typedef __attribute__((ext_vector_type(8)))  float  v8f;

static __device__ __forceinline__ v16bf join16(v8bf lo, v8bf hi) {
  v16bf r;
#pragma unroll
  for (int i = 0; i < 8; ++i) { r[i] = lo[i]; r[i + 8] = hi[i]; }
  return r;
}

static __device__ __forceinline__ void split_bf16(float v, __bf16& hi, __bf16& lo) {
  hi = (__bf16)v;
  lo = (__bf16)(v - (float)hi);
}

// ---------------------------------------------------------------------------
// B pre-pass: B[K x N] f32 row-major  ->  Bt_hi/Bt_lo [Npad x Kpad] bf16,
// zero padded, so each lane's WMMA B-fragment is contiguous.
// ---------------------------------------------------------------------------
__global__ __launch_bounds__(256) void convert_Bt_kernel(
    const float* __restrict__ B, int K, int N, int Kpad, int Npad,
    __bf16* __restrict__ BtH, __bf16* __restrict__ BtL)
{
  size_t i = (size_t)blockIdx.x * blockDim.x + threadIdx.x;
  size_t tot = (size_t)Npad * Kpad;
  if (i >= tot) return;
  int n = (int)(i / Kpad);
  int k = (int)(i % Kpad);
  float v = (n < N && k < K) ? B[(size_t)k * N + n] : 0.0f;
  __bf16 hi, lo; split_bf16(v, hi, lo);
  BtH[i] = hi; BtL[i] = lo;
}

// ---------------------------------------------------------------------------
// Generic WMMA GEMM:  C[M x N] = act( A @ B + bias )
// A is [ part1 | part2 ] row-major with optional per-row gather indices:
//   row m of A = A1[idx1 ? idx1[m] : m][0:K1]  ++  A2[idx2 ? idx2[m] : m][0:K2]
// B supplied pre-split/transposed (BtH/BtL, [Npad x Kpad] bf16).
// 3xBF16 split-precision accumulation in f32 (acc += aH*bH + aH*bL + aL*bH).
// Block = 256 threads = 8 waves; block owns a 16-row M tile; wave w owns
// N-tiles {w, w+8}. A tile staged in LDS (bank-conflict-free stride 40).
// Wave id scalarized via readfirstlane so WMMA runs under full EXEC with
// scalar s_cbranch guards (ISA: EXEC must be all 1s for WMMA).
// ---------------------------------------------------------------------------
__global__ __launch_bounds__(256) void wmma_gemm_kernel(
    const float* __restrict__ A1, const int* __restrict__ idx1, int K1,
    const float* __restrict__ A2, const int* __restrict__ idx2, int K2,
    const __bf16* __restrict__ BtH, const __bf16* __restrict__ BtL, int Kpad,
    const float* __restrict__ bias, int doRelu,
    float* __restrict__ C, int M, int N, int ldC)
{
  __shared__ __align__(16) __bf16 AsH[16 * 40];
  __shared__ __align__(16) __bf16 AsL[16 * 40];

  const int t    = (int)threadIdx.x;
  const int wave = __builtin_amdgcn_readfirstlane(t) >> 5;  // scalar wave id
  const int lane = t & 31;
  const int half = lane >> 4;   // WMMA half-wave K interleave
  const int lr   = lane & 15;
  const int m0   = (int)blockIdx.x << 4;
  const int ntiles = (N + 15) >> 4;
  const int Ktot = K1 + K2;

  int tiles[2]; int nMine = 0;            // scalar (wave is scalar)
  for (int tt = wave; tt < ntiles; tt += 8)
    if (nMine < 2) tiles[nMine++] = tt;

  v8f acc[2] = {v8f{}, v8f{}};

  // cooperative A loader: thread t loads 2 consecutive elements of the 16x32 tile
  const int aRow = t >> 4;          // 0..15
  const int aCol = (t & 15) << 1;   // 0,2,...,30
  const int am   = m0 + aRow;
  const float* rp1 = nullptr; const float* rp2 = nullptr;
  if (am < M) {
    int r1 = idx1 ? idx1[am] : am;
    rp1 = A1 + (size_t)r1 * K1;
    if (A2) { int r2 = idx2 ? idx2[am] : am; rp2 = A2 + (size_t)r2 * K2; }
  }

  for (int k0 = 0; k0 < Kpad; k0 += 32) {
    // ---- stage A tile (hi/lo bf16) into LDS, packed b32 stores ----
    {
      const int g = k0 + aCol;
      float v0 = 0.0f, v1 = 0.0f;
      if (rp1) {
        if (g + 1 < K1) {                      // fast path: both in part 1
          float2 p = *(const float2*)(rp1 + g);
          v0 = p.x; v1 = p.y;
        } else {
#pragma unroll
          for (int j = 0; j < 2; ++j) {
            int gg = g + j;
            float v = 0.0f;
            if (gg < K1) v = rp1[gg];
            else if (rp2 && gg < Ktot) v = rp2[gg - K1];
            if (j == 0) v0 = v; else v1 = v;
          }
        }
        // prefetch the A stream ahead (one prefetch per row, lane 0 of row)
        if (aCol == 0 && k0 + 256 + 32 <= K1)
          __builtin_prefetch(rp1 + k0 + 256, 0, 1);
      }
      __bf16 h0, l0, h1, l1;
      split_bf16(v0, h0, l0);
      split_bf16(v1, h1, l1);
      v2bf ph; ph[0] = h0; ph[1] = h1;
      v2bf pl; pl[0] = l0; pl[1] = l1;
      *(v2bf*)&AsH[aRow * 40 + aCol] = ph;   // ds_store_b32
      *(v2bf*)&AsL[aRow * 40 + aCol] = pl;   // ds_store_b32
    }
    __syncthreads();

    if (nMine > 0) {  // scalar branch; EXEC stays all-ones for WMMA
      const __bf16* bh = AsH + lr * 40 + half * 8;
      const __bf16* bl = AsL + lr * 40 + half * 8;
      v16bf aH = join16(*(const v8bf*)bh, *(const v8bf*)(bh + 16));
      v16bf aL = join16(*(const v8bf*)bl, *(const v8bf*)(bl + 16));
#pragma unroll
      for (int j = 0; j < 2; ++j) {
        if (j < nMine) {
          const int n0 = tiles[j] << 4;
          const size_t bo = (size_t)(n0 + lr) * Kpad + k0 + half * 8;
          v16bf bH = join16(*(const v8bf*)(BtH + bo), *(const v8bf*)(BtH + bo + 16));
          v16bf bL = join16(*(const v8bf*)(BtL + bo), *(const v8bf*)(BtL + bo + 16));
          acc[j] = __builtin_amdgcn_wmma_f32_16x16x32_bf16(false, aH, false, bH,
                                                           (short)0, acc[j], false, false);
          acc[j] = __builtin_amdgcn_wmma_f32_16x16x32_bf16(false, aH, false, bL,
                                                           (short)0, acc[j], false, false);
          acc[j] = __builtin_amdgcn_wmma_f32_16x16x32_bf16(false, aL, false, bH,
                                                           (short)0, acc[j], false, false);
        }
      }
    }
    __syncthreads();
  }

  // C layout: elem r of lane = C[half*8 + r][lr] within the 16x16 tile
#pragma unroll
  for (int j = 0; j < 2; ++j) {
    if (j < nMine) {
      const int n0 = tiles[j] << 4;
      const int col = n0 + lr;
      if (col < N) {
        const float bb = bias ? bias[col] : 0.0f;
#pragma unroll
        for (int r = 0; r < 8; ++r) {
          const int row = m0 + half * 8 + r;
          if (row < M) {
            float v = acc[j][r] + bb;
            if (doRelu) v = fmaxf(v, 0.0f);
            C[(size_t)row * ldC + col] = v;
          }
        }
      }
    }
  }
}

// ---------------------------------------------------------------------------
// Elementwise / scatter kernels
// ---------------------------------------------------------------------------
__global__ __launch_bounds__(256) void fill_f32_kernel(float* __restrict__ p, size_t n, float v) {
  size_t i = (size_t)blockIdx.x * blockDim.x + threadIdx.x;
  if (i < n) p[i] = v;
}

// edge-list SpMM scatter: Y[dst] += w_e * X[src], F multiple of 4, Y pre-zeroed
__global__ __launch_bounds__(256) void spmm_scatter_kernel(
    const float* __restrict__ X, const float* __restrict__ w,
    const int* __restrict__ src, const int* __restrict__ dst,
    int nEdges, int F, float* __restrict__ Y)
{
  size_t i = (size_t)blockIdx.x * blockDim.x + threadIdx.x;
  const int f4 = F >> 2;
  size_t tot = (size_t)nEdges * f4;
  if (i >= tot) return;
  int e = (int)(i / f4);
  int f = (int)(i % f4) << 2;
  int s = src[e], d = dst[e];
  float ww = w[e];
  float4 v = *(const float4*)(X + (size_t)s * F + f);
  float* yp = Y + (size_t)d * F + f;
  atomicAdd(yp + 0, v.x * ww);
  atomicAdd(yp + 1, v.y * ww);
  atomicAdd(yp + 2, v.z * ww);
  atomicAdd(yp + 3, v.w * ww);
}

__global__ __launch_bounds__(256) void bias_act_kernel(
    float* __restrict__ Y, const float* __restrict__ b, size_t total, int F, int doRelu)
{
  size_t i = (size_t)blockIdx.x * blockDim.x + threadIdx.x;
  if (i >= total) return;
  int f = (int)(i % F);
  float v = Y[i] + b[f];
  if (doRelu) v = fmaxf(v, 0.0f);
  Y[i] = v;
}

__global__ __launch_bounds__(256) void scatter_sum_feat_kernel(
    const float* __restrict__ H, const int* __restrict__ seg,
    int Mrows, int F, float* __restrict__ S)
{
  size_t i = (size_t)blockIdx.x * blockDim.x + threadIdx.x;
  const int f4 = F >> 2;
  size_t tot = (size_t)Mrows * f4;
  if (i >= tot) return;
  int m = (int)(i / f4);
  int f = (int)(i % f4) << 2;
  int e = seg[m];
  float4 v = *(const float4*)(H + (size_t)m * F + f);
  float* sp = S + (size_t)e * F + f;
  atomicAdd(sp + 0, v.x);
  atomicAdd(sp + 1, v.y);
  atomicAdd(sp + 2, v.z);
  atomicAdd(sp + 3, v.w);
}

__global__ __launch_bounds__(256) void scatter_count_kernel(
    const int* __restrict__ seg, int Mrows, float* __restrict__ cnt)
{
  int m = blockIdx.x * blockDim.x + threadIdx.x;
  if (m < Mrows) atomicAdd(&cnt[seg[m]], 1.0f);
}

__global__ __launch_bounds__(256) void mean_div_kernel(
    float* __restrict__ S, const float* __restrict__ cnt, int E, int F)
{
  size_t i = (size_t)blockIdx.x * blockDim.x + threadIdx.x;
  size_t tot = (size_t)E * F;
  if (i >= tot) return;
  int e = (int)(i / F);
  S[i] = S[i] / fmaxf(cnt[e], 1.0f);
}

// wave-per-node dot product: score[m] = [h2[m] | mean[bx[m]]] . Wemb + bemb
__global__ __launch_bounds__(256) void score_kernel(
    const float* __restrict__ h2, const float* __restrict__ mean,
    const int* __restrict__ bx, const float* __restrict__ Wemb,
    const float* __restrict__ bemb, int Mrows, float* __restrict__ score)
{
  int wid  = (int)(((size_t)blockIdx.x * blockDim.x + threadIdx.x) >> 5);
  int lane = threadIdx.x & 31;
  if (wid >= Mrows) return;
  int e = bx[wid];
  float s = 0.0f;
  for (int k = lane; k < 400; k += 32) {
    float v = (k < 200) ? h2[(size_t)wid * 200 + k] : mean[(size_t)e * 200 + (k - 200)];
    s += v * Wemb[k];
  }
#pragma unroll
  for (int off = 16; off > 0; off >>= 1) s += __shfl_xor(s, off, 32);
  if (lane == 0) score[wid] = s + bemb[0];
}

// monotonic float<->uint encoding for atomicMax-based segment max
static __device__ __forceinline__ unsigned enc_f(float f) {
  unsigned u = __float_as_uint(f);
  return (u & 0x80000000u) ? ~u : (u | 0x80000000u);
}
static __device__ __forceinline__ float dec_f(unsigned u) {
  unsigned v = (u & 0x80000000u) ? (u & 0x7FFFFFFFu) : ~u;
  return __uint_as_float(v);
}

__global__ __launch_bounds__(256) void seg_max_kernel(
    const float* __restrict__ score, const int* __restrict__ bx,
    int Mrows, unsigned* __restrict__ smax)
{
  int m = blockIdx.x * blockDim.x + threadIdx.x;
  if (m < Mrows) atomicMax(&smax[bx[m]], enc_f(score[m]));
}

__global__ __launch_bounds__(256) void exp_sum_kernel(
    float* __restrict__ score, const int* __restrict__ bx,
    const unsigned* __restrict__ smax, int Mrows, float* __restrict__ esum)
{
  int m = blockIdx.x * blockDim.x + threadIdx.x;
  if (m >= Mrows) return;
  int e = bx[m];
  float ex = __expf(score[m] - dec_f(smax[e]));
  score[m] = ex;                 // e_i stored in place
  atomicAdd(&esum[e], ex);
}

// newsum[bx[m]][f] += (e_m / esum[bx[m]]) * emb[m][f],  emb = [h2 | mean[bx]]
__global__ __launch_bounds__(256) void weighted_scatter_kernel(
    const float* __restrict__ escore, const float* __restrict__ esum,
    const int* __restrict__ bx, const float* __restrict__ h2,
    const float* __restrict__ mean, int Mrows, float* __restrict__ newsum)
{
  size_t i = (size_t)blockIdx.x * blockDim.x + threadIdx.x;
  size_t tot = (size_t)Mrows * 400;
  if (i >= tot) return;
  int m = (int)(i / 400);
  int f = (int)(i % 400);
  int e = bx[m];
  float z = escore[m] / esum[e];
  float v = (f < 200) ? h2[(size_t)m * 200 + f] : mean[(size_t)e * 200 + (f - 200)];
  atomicAdd(&newsum[(size_t)e * 400 + f], z * v);
}

// ---------------------------------------------------------------------------
// Host launcher
// ---------------------------------------------------------------------------
static inline int cdiv_ll(long long a, long long b) { return (int)((a + b - 1) / b); }
static inline int kpad32(int k) { return ((k + 31) / 32) * 32; }
static inline int npad16(int n) { return ((n + 15) / 16) * 16; }

extern "C" void kernel_launch(void* const* d_in, const int* in_sizes, int n_in,
                              void* d_out, int out_size, void* d_ws, size_t ws_size,
                              hipStream_t stream)
{
  (void)in_sizes; (void)n_in; (void)out_size; (void)ws_size;
  const float* img_feat    = (const float*)d_in[0];
  const float* text_feat   = (const float*)d_in[1];
  const float* rel_emb     = (const float*)d_in[2];
  const float* W_img_text  = (const float*)d_in[3];
  const float* b_img_text  = (const float*)d_in[4];
  const float* W_rel_input = (const float*)d_in[5];
  const float* b_rel_input = (const float*)d_in[6];
  const float* W_gc1       = (const float*)d_in[7];
  const float* b_gc1       = (const float*)d_in[8];
  const float* W_gc2       = (const float*)d_in[9];
  const float* b_gc2       = (const float*)d_in[10];
  const float* W_emb       = (const float*)d_in[11];
  const float* b_emb       = (const float*)d_in[12];
  const float* W_emb_out   = (const float*)d_in[13];
  const float* b_emb_out   = (const float*)d_in[14];
  const float* W_rel       = (const float*)d_in[15];
  const float* b_rel       = (const float*)d_in[16];
  const float* edge_w      = (const float*)d_in[17];
  const int*   b_x         = (const int*)d_in[18];
  const int*   b_graph     = (const int*)d_in[19];
  const int*   edge_src    = (const int*)d_in[20];
  const int*   edge_dst    = (const int*)d_in[21];

  // ---- workspace carve-up ----
  char* base = (char*)d_ws;
  size_t off = 0;
  auto alloc = [&](size_t bytes) -> char* {
    off = (off + 255) & ~(size_t)255;
    char* p = base + off;
    off += bytes;
    return p;
  };
  float*    emf     = (float*)alloc((size_t)N_ENT * 100 * 4);
  float*    relin   = (float*)alloc((size_t)N_REL * 100 * 4);
  const size_t BT_ELEMS = (size_t)256 * 4416;           // max Npad*Kpad over all GEMMs
  __bf16*   BtH     = (__bf16*)alloc(BT_ELEMS * 2);
  __bf16*   BtL     = (__bf16*)alloc(BT_ELEMS * 2);
  float*    xw      = (float*)alloc((size_t)N_NODES * 256 * 4);  // reused for XW2
  float*    h1      = (float*)alloc((size_t)N_NODES * 256 * 4);
  float*    h2      = (float*)alloc((size_t)N_NODES * 200 * 4);
  float*    meansum = (float*)alloc((size_t)N_ENT * 200 * 4);
  float*    cnt     = (float*)alloc((size_t)N_ENT * 4);
  float*    score   = (float*)alloc((size_t)N_NODES * 4);
  unsigned* smax    = (unsigned*)alloc((size_t)N_ENT * 4);
  float*    esum    = (float*)alloc((size_t)N_ENT * 4);
  float*    newsum  = (float*)alloc((size_t)N_ENT * 400 * 4);

  float* out_new = (float*)d_out;                       // [N_ENT x 200]
  float* out_rel = (float*)d_out + (size_t)N_ENT * 200; // [N_REL x 200]

  dim3 blk(256);

  auto conv = [&](const float* B, int K, int N) {
    int Kp = kpad32(K), Np = npad16(N);
    size_t tot = (size_t)Np * Kp;
    convert_Bt_kernel<<<cdiv_ll((long long)tot, 256), blk, 0, stream>>>(B, K, N, Kp, Np, BtH, BtL);
  };
  auto gemm = [&](const float* A1, const int* i1, int K1,
                  const float* A2, const int* i2, int K2,
                  const float* bias, int relu, float* C, int M, int N) {
    int Kp = kpad32(K1 + K2);
    wmma_gemm_kernel<<<cdiv_ll(M, 16), blk, 0, stream>>>(
        A1, i1, K1, A2, i2, K2, BtH, BtL, Kp, bias, relu, C, M, N, N);
  };
  auto fill0 = [&](float* p, size_t n) {
    fill_f32_kernel<<<cdiv_ll((long long)n, 256), blk, 0, stream>>>(p, n, 0.0f);
  };

  // 1) entity encoder: emf = [img|text] @ W_img_text + b        [N_ENT x 100]
  conv(W_img_text, 4396, 100);
  gemm(img_feat, nullptr, 4096, text_feat, nullptr, 300, b_img_text, 0, emf, N_ENT, 100);

  // 2) relation context encoder: relin = relu(rel_emb @ W_rel_input + b)  [N_REL x 100]
  conv(W_rel_input, 300, 100);
  gemm(rel_emb, nullptr, 300, nullptr, nullptr, 0, b_rel_input, 1, relin, N_REL, 100);

  // 3) gc1: XW1 = [emf[b_x] | relin[b_graph]] @ W_gc1 (gather fused into GEMM)
  conv(W_gc1, 200, 256);
  gemm(emf, b_x, 100, relin, b_graph, 100, nullptr, 0, xw, N_NODES, 256);

  // 4) h1 = relu(spmm(XW1) + b_gc1)
  fill0(h1, (size_t)N_NODES * 256);
  spmm_scatter_kernel<<<cdiv_ll((long long)N_EDGES * 64, 256), blk, 0, stream>>>(
      xw, edge_w, edge_src, edge_dst, N_EDGES, 256, h1);
  bias_act_kernel<<<cdiv_ll((long long)N_NODES * 256, 256), blk, 0, stream>>>(
      h1, b_gc1, (size_t)N_NODES * 256, 256, 1);

  // 5) gc2: XW2 = h1 @ W_gc2 ; h2 = spmm(XW2) + b_gc2
  conv(W_gc2, 256, 200);
  gemm(h1, nullptr, 256, nullptr, nullptr, 0, nullptr, 0, xw, N_NODES, 200);
  fill0(h2, (size_t)N_NODES * 200);
  spmm_scatter_kernel<<<cdiv_ll((long long)N_EDGES * 50, 256), blk, 0, stream>>>(
      xw, edge_w, edge_src, edge_dst, N_EDGES, 200, h2);
  bias_act_kernel<<<cdiv_ll((long long)N_NODES * 200, 256), blk, 0, stream>>>(
      h2, b_gc2, (size_t)N_NODES * 200, 200, 0);

  // 6) scatter_mean over b_x
  fill0(meansum, (size_t)N_ENT * 200);
  fill0(cnt, N_ENT);
  scatter_sum_feat_kernel<<<cdiv_ll((long long)N_NODES * 50, 256), blk, 0, stream>>>(
      h2, b_x, N_NODES, 200, meansum);
  scatter_count_kernel<<<cdiv_ll(N_NODES, 256), blk, 0, stream>>>(b_x, N_NODES, cnt);
  mean_div_kernel<<<cdiv_ll((long long)N_ENT * 200, 256), blk, 0, stream>>>(
      meansum, cnt, N_ENT, 200);   // meansum now holds mean

  // 7) score = emb @ W_emb + b_emb   (emb recomputed on the fly)
  score_kernel<<<cdiv_ll((long long)N_NODES * 32, 256), blk, 0, stream>>>(
      h2, meansum, b_x, W_emb, b_emb, N_NODES, score);

  // 8) scatter softmax over b_x
  fill0((float*)smax, N_ENT);   // uint 0 == encoding floor
  seg_max_kernel<<<cdiv_ll(N_NODES, 256), blk, 0, stream>>>(score, b_x, N_NODES, smax);
  fill0(esum, N_ENT);
  exp_sum_kernel<<<cdiv_ll(N_NODES, 256), blk, 0, stream>>>(score, b_x, smax, N_NODES, esum);

  // 9) newsum = seg_sum(z * emb)   [N_ENT x 400]
  fill0(newsum, (size_t)N_ENT * 400);
  weighted_scatter_kernel<<<cdiv_ll((long long)N_NODES * 400, 256), blk, 0, stream>>>(
      score, esum, b_x, h2, meansum, N_NODES, newsum);

  // 10) new_out = newsum @ W_emb_out + b_emb_out   -> d_out[0 : N_ENT*200]
  conv(W_emb_out, 400, 200);
  gemm(newsum, nullptr, 400, nullptr, nullptr, 0, b_emb_out, 0, out_new, N_ENT, 200);

  // 11) rel_context = relu(rel_emb @ W_rel + b_rel) -> d_out[N_ENT*200 : ]
  conv(W_rel, 300, 200);
  gemm(rel_emb, nullptr, 300, nullptr, nullptr, 0, b_rel, 1, out_rel, N_REL, 200);
}
